// Scaled_Dot_Product_Self_Attention_53077205844765
// MI455X (gfx1250) — matrix operand order repaired and checked by reading, not compiled, not run on hardware
//
#include <hip/hip_runtime.h>

// ---------------------------------------------------------------------------
// Scaled dot-product self-attention for MI455X (gfx1250, wave32, WMMA).
// Pipeline (all bf16 WMMA with f32 accumulate):
//   1) convert x, Wq, Wk, Wv  f32 -> bf16
//   2) NT-GEMM: Q = (x/sqrt(D)) Wq^T, K = x Wk^T, Vt = (x Wv^T)^T   (bf16)
//   3) NT-GEMM: S[b] = Q[b] K[b]^T                                  (bf16)
//   4) row softmax over S (f32 math, in place, bf16 out)
//   5) NT-GEMM: out[b] = S[b] Vt[b]^T                               (f32)
// Every GEMM is NT => both WMMA fragments are contiguous row chunks.
// Fragment layouts per CDNA5 ISA §7.12.2:
//   A (16x32 bf16): lane r (0..15) + half h: K = {8h..8h+7, 16+8h..16+8h+7}
//   B (32x16 bf16): lane r + half h: column r, K = {8h..8h+7, 16+8h..16+8h+7} (same as A)
//   C/D (16x16 f32): VGPR i -> M = i + 8h, N = lane%16
// GEMM tiling: 8 waves (2M x 4N), wave tile 64x32 (4x2 fragments) =>
// 12 b128 loads per 8 WMMAs per k-step.
// ---------------------------------------------------------------------------

typedef __attribute__((ext_vector_type(16))) __bf16 v16bf;
typedef __attribute__((ext_vector_type(8)))  float  v8f;
typedef __attribute__((ext_vector_type(4)))  int    i32x4;
typedef __attribute__((ext_vector_type(4)))  float  f32x4;

union FragBF {
    v16bf v;
    i32x4 q[2];
};

// ---------------------------------------------------------------------------
// f32 -> bf16 conversion, 4 elements/thread (n must be a multiple of 4)
// ---------------------------------------------------------------------------
__global__ __launch_bounds__(256) void cvt_f32_to_bf16(const float* __restrict__ src,
                                                       __bf16* __restrict__ dst,
                                                       int n) {
    int i = (blockIdx.x * 256 + threadIdx.x) * 4;
    if (i < n) {
        f32x4 v = *(const f32x4*)(src + i);
        dst[i + 0] = (__bf16)v[0];
        dst[i + 1] = (__bf16)v[1];
        dst[i + 2] = (__bf16)v[2];
        dst[i + 3] = (__bf16)v[3];
    }
}

// ---------------------------------------------------------------------------
// NT GEMM: C[m,n] = alpha * sum_k A[m,k] * B[n,k]   (A: MxK, B: NxK row-major)
// Block = 256 threads = 8 waves arranged 2(M) x 4(N); wave tile 64x32
// (4x2 fragments of 16x16), block tile 128x128, K-step 32.
// grid = (N/128, M/128, batches)
// MODE 0: store bf16, row-major         (ldc = row stride)
// MODE 1: store bf16 TRANSPOSED with per-2048-row batch split (for V^T)
// MODE 2: store f32, row-major
// ---------------------------------------------------------------------------
template <int MODE>
__global__ __launch_bounds__(256) void gemm_nt_bf16_wmma(
    const __bf16* __restrict__ A, const __bf16* __restrict__ B,
    void* __restrict__ Cv,
    int lda, int ldb, int ldc, int Kdim,
    long long strideA, long long strideB, long long strideC,
    float alpha) {
    const int lane = threadIdx.x & 31;
    const int wave = threadIdx.x >> 5;
    const int wm   = wave & 1;   // wave row    0..1
    const int wn   = wave >> 1;  // wave column 0..3

    const long long zb = blockIdx.z;
    A += zb * strideA;
    B += zb * strideB;

    const int m0 = blockIdx.y * 128 + wm * 64;
    const int n0 = blockIdx.x * 128 + wn * 32;

    const int r   = lane & 15;
    const int hh  = lane >> 4;   // lane half
    const int akh = hh << 3;     // A: K offset 0 or 8 (interleaved 8-chunks)
    const int bkh = hh << 3;     // B: K offset 0 or 8 (interleaved 8-chunks)

    const __bf16* ap[4];
    const __bf16* bp[2];
#pragma unroll
    for (int mi = 0; mi < 4; ++mi)
        ap[mi] = A + (size_t)(m0 + mi * 16 + r) * lda + akh;
#pragma unroll
    for (int ni = 0; ni < 2; ++ni)
        bp[ni] = B + (size_t)(n0 + ni * 16 + r) * ldb + bkh;

    v8f acc[4][2];
#pragma unroll
    for (int mi = 0; mi < 4; ++mi)
#pragma unroll
        for (int ni = 0; ni < 2; ++ni) acc[mi][ni] = {};

    for (int k0 = 0; k0 < Kdim; k0 += 32) {
        FragBF fa[4], fb[2];
#pragma unroll
        for (int mi = 0; mi < 4; ++mi) {
            fa[mi].q[0] = *(const i32x4*)(ap[mi] + k0);       // K = akh + 0..7
            fa[mi].q[1] = *(const i32x4*)(ap[mi] + k0 + 16);  // K = akh + 16..23
        }
#pragma unroll
        for (int ni = 0; ni < 2; ++ni) {
            fb[ni].q[0] = *(const i32x4*)(bp[ni] + k0);       // K = bkh + 0..7
            fb[ni].q[1] = *(const i32x4*)(bp[ni] + k0 + 16);  // K = bkh + 16..23
        }
#pragma unroll
        for (int mi = 0; mi < 4; ++mi)
#pragma unroll
            for (int ni = 0; ni < 2; ++ni)
                acc[mi][ni] = __builtin_amdgcn_wmma_f32_16x16x32_bf16(
                    false, fa[mi].v, false, fb[ni].v, (short)0, acc[mi][ni], false, false);
    }

    // C/D fragment layout: VGPR i -> M = i + 8*(lane>=16), N = lane%16
    const int cn = lane & 15;
    const int mh = hh << 3;

#pragma unroll
    for (int mi = 0; mi < 4; ++mi) {
#pragma unroll
        for (int ni = 0; ni < 2; ++ni) {
#pragma unroll
            for (int i = 0; i < 8; ++i) {
                const float val = alpha * acc[mi][ni][i];
                const int mm = m0 + mi * 16 + mh + i;
                const int nn = n0 + ni * 16 + cn;
                if (MODE == 0) {
                    ((__bf16*)Cv)[(size_t)zb * strideC + (size_t)mm * ldc + nn] =
                        (__bf16)val;
                } else if (MODE == 1) {
                    const int bb = mm >> 11;   // batch  = m / 2048
                    const int ml = mm & 2047;  // m within batch
                    ((__bf16*)Cv)[(size_t)bb * strideC + (size_t)nn * ldc + ml] =
                        (__bf16)val;
                } else {
                    ((float*)Cv)[(size_t)zb * strideC + (size_t)mm * ldc + nn] = val;
                }
            }
        }
    }
}

// ---------------------------------------------------------------------------
// Row softmax, in place on bf16 buffer. One block per row, 2048 cols,
// 8 elements/thread, f32 math with LDS tree reductions.
// ---------------------------------------------------------------------------
__global__ __launch_bounds__(256) void softmax_rows_bf16(__bf16* __restrict__ S) {
    __shared__ float red[256];
    const int t = threadIdx.x;
    __bf16* p = S + (size_t)blockIdx.x * 2048;

    float v[8];
    float mx = -3.0e38f;
#pragma unroll
    for (int j = 0; j < 8; ++j) {
        v[j] = (float)p[t + j * 256];
        mx = fmaxf(mx, v[j]);
    }
    red[t] = mx;
    __syncthreads();
    for (int s = 128; s > 0; s >>= 1) {
        if (t < s) red[t] = fmaxf(red[t], red[t + s]);
        __syncthreads();
    }
    mx = red[0];
    __syncthreads();

    float sum = 0.0f;
#pragma unroll
    for (int j = 0; j < 8; ++j) {
        v[j] = __expf(v[j] - mx);
        sum += v[j];
    }
    red[t] = sum;
    __syncthreads();
    for (int s = 128; s > 0; s >>= 1) {
        if (t < s) red[t] += red[t + s];
        __syncthreads();
    }
    const float inv = 1.0f / red[0];

#pragma unroll
    for (int j = 0; j < 8; ++j) p[t + j * 256] = (__bf16)(v[j] * inv);
}

// ---------------------------------------------------------------------------
extern "C" void kernel_launch(void* const* d_in, const int* in_sizes, int n_in,
                              void* d_out, int out_size, void* d_ws, size_t ws_size,
                              hipStream_t stream) {
    (void)in_sizes; (void)n_in; (void)out_size; (void)ws_size;

    constexpr int B = 4, N = 2048, D = 1024;
    constexpr size_t XE = (size_t)B * N * D;  // 8388608
    constexpr size_t WE = (size_t)D * D;      // 1048576
    constexpr size_t SE = (size_t)B * N * N;  // 16777216

    const float* x  = (const float*)d_in[0];
    const float* Wq = (const float*)d_in[1];
    const float* Wk = (const float*)d_in[2];
    const float* Wv = (const float*)d_in[3];
    float* out = (float*)d_out;

    char* ws = (char*)d_ws;
    __bf16* X16  = (__bf16*)ws; ws += XE * 2;
    __bf16* Wq16 = (__bf16*)ws; ws += WE * 2;
    __bf16* Wk16 = (__bf16*)ws; ws += WE * 2;
    __bf16* Wv16 = (__bf16*)ws; ws += WE * 2;
    __bf16* Q    = (__bf16*)ws; ws += XE * 2;
    __bf16* Kb   = (__bf16*)ws; ws += XE * 2;
    __bf16* Vt   = (__bf16*)ws; ws += XE * 2;  // per batch: [D][N]
    __bf16* S    = (__bf16*)ws; ws += SE * 2;  // per batch: [N][N]

    // 1) convert to bf16
    cvt_f32_to_bf16<<<dim3((unsigned)(XE / 4 / 256)), 256, 0, stream>>>(x,  X16, (int)XE);
    cvt_f32_to_bf16<<<dim3((unsigned)(WE / 4 / 256)), 256, 0, stream>>>(Wq, Wq16, (int)WE);
    cvt_f32_to_bf16<<<dim3((unsigned)(WE / 4 / 256)), 256, 0, stream>>>(Wk, Wk16, (int)WE);
    cvt_f32_to_bf16<<<dim3((unsigned)(WE / 4 / 256)), 256, 0, stream>>>(Wv, Wv16, (int)WE);

    // 2) projections: M = B*N = 8192 (batch-independent), N = D, K = D
    const dim3 gProj(D / 128, (B * N) / 128, 1);
    // Q = (x Wq^T) / sqrt(D)   (scale folded into Q)
    gemm_nt_bf16_wmma<0><<<gProj, 256, 0, stream>>>(X16, Wq16, Q, D, D, D, D,
                                                    0, 0, 0, 0.03125f);
    gemm_nt_bf16_wmma<0><<<gProj, 256, 0, stream>>>(X16, Wk16, Kb, D, D, D, D,
                                                    0, 0, 0, 1.0f);
    // V stored transposed per batch: Vt[b][d][m], ldc = N, batch stride = D*N
    gemm_nt_bf16_wmma<1><<<gProj, 256, 0, stream>>>(X16, Wv16, Vt, D, D, N, D,
                                                    0, 0, (long long)D * N, 1.0f);

    // 3) energy: S[b] = Q[b] K[b]^T   (M=N=2048, K=D, per batch)
    const dim3 gEng(N / 128, N / 128, B);
    gemm_nt_bf16_wmma<0><<<gEng, 256, 0, stream>>>(Q, Kb, S, D, D, N, D,
                                                   (long long)N * D, (long long)N * D,
                                                   (long long)N * N, 1.0f);

    // 4) softmax over rows of S (B*N rows of length N)
    softmax_rows_bf16<<<dim3(B * N), 256, 0, stream>>>(S);

    // 5) out[b] = S[b] Vt[b]^T  (M=2048, N=D, K=2048, f32 output)
    const dim3 gOut(D / 128, N / 128, B);
    gemm_nt_bf16_wmma<2><<<gOut, 256, 0, stream>>>(S, Vt, out, N, N, D, N,
                                                   (long long)N * N, (long long)D * N,
                                                   (long long)N * D, 1.0f);
}
